// dwt_concat3_38371237822843
// MI455X (gfx1250) — compile-verified, matrix operand order checked by reading, not run on hardware
//
#include <hip/hip_runtime.h>
#include <hip/hip_bf16.h>

// ---------------------------------------------------------------------------
// Fused DWT(x1,x2,x3) -> combine -> IDWT -> concat -> 1x1 conv (GEMM, bf16 WMMA)
// B=4, C=96, H=256, W=256, COUT=96, K=3C=288
// Tile per workgroup: 2 rows x 64 cols (Haar-block aligned), 256 threads.
// ---------------------------------------------------------------------------

#define B_    4
#define C_    96
#define H_    256
#define W_    256
#define COUT_ 96
#define K_    288         // 3*C
#define TROWS 2
#define TCOLS 64
#define TSPAT (TROWS*TCOLS)   // 128 spatial positions / WG
#define PITCH 296             // bf16 elements per LDS row (288 + 8 pad) -> 148 dwords

typedef __attribute__((ext_vector_type(16))) __bf16 v16bf;
typedef __attribute__((ext_vector_type(8)))  __bf16 v8bf;
typedef __attribute__((ext_vector_type(8)))  float  v8f;

__device__ __forceinline__ unsigned short f32_to_bf16(float f) {
    unsigned int u = __float_as_uint(f);
    unsigned int r = u + 0x7FFFu + ((u >> 16) & 1u);   // round-to-nearest-even
    return (unsigned short)(r >> 16);
}

__device__ __forceinline__ float ffm(float a, float b) {
    return (fabsf(a) >= fabsf(b)) ? a : b;
}

__device__ __forceinline__ v16bf frag_from_lds(const unsigned short* row,
                                               int c_lo, int c_hi) {
    // two ds_load_b128 of 8 consecutive bf16 channels each; matches the
    // CDNA5 16-bit A/B fragment VGPR layout (ISA 7.12.2).
    v8bf lo = *(const v8bf*)(row + c_lo);
    v8bf hi = *(const v8bf*)(row + c_hi);
    return __builtin_shufflevector(lo, hi, 0,1,2,3,4,5,6,7,
                                           8,9,10,11,12,13,14,15);
}

__global__ __launch_bounds__(256) void dwt_fuse_gemm_kernel(
    const float* __restrict__ x1,
    const float* __restrict__ x2,
    const float* __restrict__ x3,
    const float* __restrict__ w_out,   // [96][288] row-major
    float* __restrict__ out)           // [4][96][256][256]
{
    extern __shared__ __align__(16) unsigned char smem_raw[];
    unsigned short* y_lds = (unsigned short*)smem_raw;      // [128][PITCH] bf16
    unsigned short* w_lds = y_lds + TSPAT * PITCH;          // [96][PITCH]  bf16

    // ---- tile coordinates -------------------------------------------------
    const int tile = blockIdx.x;              // 0..2047
    const int b    = tile >> 9;               // 512 tiles per image
    const int rem  = tile & 511;
    const int h0   = (rem >> 2) * 2;          // row-pair base
    const int w0   = (rem & 3) * TCOLS;       // column base

    const int t = threadIdx.x;                // 0..255

    // ---- stage weights into LDS as bf16 ----------------------------------
    for (int i = t; i < COUT_ * K_; i += 256) {
        int o = i / K_, c = i - o * K_;
        w_lds[o * PITCH + c] = f32_to_bf16(w_out[i]);
    }

    // ---- stage Y = [x_rec | x2 | x3] into LDS as bf16 ---------------------
    // 32 Haar blocks (2x2) per tile; 8 channel-slots of 12 channels.
    {
        const int bk    = t & 31;             // block index -> cols 2bk,2bk+1
        const int cslot = t >> 5;             // 0..7
        const int wblk  = bk * 2;

        for (int j = 0; j < 12; ++j) {
            const int c = cslot * 12 + j;
            const long base = (((long)(b * C_ + c) * H_ + h0) * W_) + w0 + wblk;

            float2 a0 = *(const float2*)(x1 + base);
            float2 a1 = *(const float2*)(x1 + base + W_);
            float2 b0 = *(const float2*)(x2 + base);
            float2 b1 = *(const float2*)(x2 + base + W_);
            float2 c0 = *(const float2*)(x3 + base);
            float2 c1 = *(const float2*)(x3 + base + W_);

            // Haar per 2x2 block (everything scaled by 0.5 as in reference)
            float A1 = 0.5f*( a0.x + a1.x + a0.y + a1.y);
            float B1 = 0.5f*(-a0.x - a1.x + a0.y + a1.y);
            float C1 = 0.5f*(-a0.x + a1.x - a0.y + a1.y);
            float D1 = 0.5f*( a0.x - a1.x - a0.y + a1.y);

            float A2 = 0.5f*( b0.x + b1.x + b0.y + b1.y);
            float B2 = 0.5f*(-b0.x - b1.x + b0.y + b1.y);
            float C2 = 0.5f*(-b0.x + b1.x - b0.y + b1.y);
            float D2 = 0.5f*( b0.x - b1.x - b0.y + b1.y);

            float A3 = 0.5f*( c0.x + c1.x + c0.y + c1.y);
            float B3 = 0.5f*(-c0.x - c1.x + c0.y + c1.y);
            float C3 = 0.5f*(-c0.x + c1.x - c0.y + c1.y);
            float D3 = 0.5f*( c0.x - c1.x - c0.y + c1.y);

            float Am = (A1 + A2 + A3) * (1.0f / 3.0f);
            float Bm = ffm(ffm(B1, B2), B3);
            float Cm = ffm(ffm(C1, C2), C3);
            float Dm = ffm(ffm(D1, D2), D3);

            // IDWT with the *2 interleave scaling folded in (q*2 = 0.5)
            float r00 = 0.5f * (Am - Bm - Cm + Dm);   // even row, even col
            float r10 = 0.5f * (Am - Bm + Cm - Dm);   // odd  row, even col
            float r01 = 0.5f * (Am + Bm - Cm - Dm);   // even row, odd  col
            float r11 = 0.5f * (Am + Bm + Cm + Dm);   // odd  row, odd  col

            unsigned short* p00 = y_lds + (0*TCOLS + wblk    ) * PITCH;
            unsigned short* p01 = y_lds + (0*TCOLS + wblk + 1) * PITCH;
            unsigned short* p10 = y_lds + (1*TCOLS + wblk    ) * PITCH;
            unsigned short* p11 = y_lds + (1*TCOLS + wblk + 1) * PITCH;

            p00[c] = f32_to_bf16(r00);  p01[c] = f32_to_bf16(r01);
            p10[c] = f32_to_bf16(r10);  p11[c] = f32_to_bf16(r11);

            p00[96 + c] = f32_to_bf16(b0.x);  p01[96 + c] = f32_to_bf16(b0.y);
            p10[96 + c] = f32_to_bf16(b1.x);  p11[96 + c] = f32_to_bf16(b1.y);

            p00[192 + c] = f32_to_bf16(c0.x); p01[192 + c] = f32_to_bf16(c0.y);
            p10[192 + c] = f32_to_bf16(c1.x); p11[192 + c] = f32_to_bf16(c1.y);
        }
    }

    __syncthreads();

    // ---- WMMA GEMM: out[16o x 16s] += W[16o x 32k] * Y[32k x 16s] ---------
    const int wv   = t >> 5;      // wave id, one 16-wide spatial tile each
    const int lane = t & 31;
    const int half = lane >> 4;
    const int l16  = lane & 15;

    const unsigned short* yrow = y_lds + (wv * 16 + l16) * PITCH;

    v8f acc[6];
#pragma unroll
    for (int ot = 0; ot < 6; ++ot) acc[ot] = (v8f){0,0,0,0,0,0,0,0};

    for (int kc = 0; kc < 9; ++kc) {
        const int c_lo = kc * 32 + half * 8;
        const int c_hi = c_lo + 16;

        v16bf bFrag = frag_from_lds(yrow, c_lo, c_hi);

#pragma unroll
        for (int ot = 0; ot < 6; ++ot) {
            const unsigned short* wrow = w_lds + (ot * 16 + l16) * PITCH;
            v16bf aFrag = frag_from_lds(wrow, c_lo, c_hi);
            acc[ot] = __builtin_amdgcn_wmma_f32_16x16x32_bf16(
                false, aFrag, false, bFrag, (short)0, acc[ot], false, false);
        }
    }

    // ---- store: D[m][n] -> m = o = ot*16 + v + 8*half, n = spatial col ----
    const int h    = h0 + (wv >> 2);
    const int wcol = w0 + (wv & 3) * 16 + l16;

#pragma unroll
    for (int ot = 0; ot < 6; ++ot) {
#pragma unroll
        for (int v = 0; v < 8; ++v) {
            const int o = ot * 16 + v + 8 * half;
            out[(((long)(b * COUT_ + o) * H_ + h) * W_) + wcol] = acc[ot][v];
        }
    }
}

extern "C" void kernel_launch(void* const* d_in, const int* in_sizes, int n_in,
                              void* d_out, int out_size, void* d_ws, size_t ws_size,
                              hipStream_t stream) {
    (void)in_sizes; (void)n_in; (void)out_size; (void)d_ws; (void)ws_size;

    const float* x1    = (const float*)d_in[0];
    const float* x2    = (const float*)d_in[1];
    const float* x3    = (const float*)d_in[2];
    const float* w_out = (const float*)d_in[3];
    float*       out   = (float*)d_out;

    const int tiles = B_ * (H_ / TROWS) * (W_ / TCOLS);          // 2048
    const size_t lds_bytes = (size_t)(TSPAT + COUT_) * PITCH * 2; // 132,608 B

    dwt_fuse_gemm_kernel<<<tiles, 256, lds_bytes, stream>>>(x1, x2, x3, w_out, out);
}
